// CGNN_20401094656217
// MI455X (gfx1250) — matrix operand order repaired and checked by reading, not compiled, not run on hardware
//
#include <hip/hip_runtime.h>

// Problem constants (match reference)
#define NN   30000
#define EE   240000
#define IND  512
#define DD   128
#define OUTD 64
#define VV   2
#define BLKN 2
#define HH   2
#define DMG  64
#define NEG_ATT 0.2f
#define NEG_OUT 0.01f
#define CONCATK (DD + VV*HH*DD)   // 640

typedef __attribute__((ext_vector_type(16))) _Float16 v16h;
typedef __attribute__((ext_vector_type(8)))  float    v8f;

// ---------- helpers ----------
__device__ __forceinline__ unsigned f2o(float f) {
  unsigned u = __float_as_uint(f);
  return (u & 0x80000000u) ? ~u : (u | 0x80000000u);
}
__device__ __forceinline__ float o2f(unsigned u) {
  return (u & 0x80000000u) ? __uint_as_float(u & 0x7fffffffu) : __uint_as_float(~u);
}
// ordered encoding of -inf
#define ORD_NEG_INF 0x007FFFFFu

__global__ __launch_bounds__(256) void fill_u32(unsigned* __restrict__ p, unsigned v, int n) {
  for (int i = blockIdx.x * blockDim.x + threadIdx.x; i < n; i += gridDim.x * blockDim.x)
    p[i] = v;
}

__global__ __launch_bounds__(256) void cvt_f32_f16(const float* __restrict__ in,
                                                   _Float16* __restrict__ out, int n) {
  for (int i = blockIdx.x * blockDim.x + threadIdx.x; i < n; i += gridDim.x * blockDim.x)
    out[i] = (_Float16)in[i];
}

// transpose + convert: in is [K, C] f32 row-major, out is [C, K] f16 row-major
__global__ __launch_bounds__(256) void tconv(const float* __restrict__ in,
                                             _Float16* __restrict__ out, int K, int C) {
  int n = K * C;
  for (int i = blockIdx.x * blockDim.x + threadIdx.x; i < n; i += gridDim.x * blockDim.x) {
    int c = i % C, k = i / C;
    out[(size_t)c * K + k] = (_Float16)in[i];
  }
}

// ---------- WMMA GEMM: C[M,Ncols] = A[M,K](f16,row) * Bt[Ncols,K](f16,row = B col-major) ----------
// One wave computes a 16x64 output strip (4 x 16x16 WMMA tiles): the A fragment is loaded once
// per K-step and feeds 4 v_wmma_f32_16x16x32_f16, giving 4 independent accumulator chains.
// Requires Ncols % 64 == 0 (true for 128/256/64 here).
// Per-lane fragment addressing (CDNA5 wave32):
//  A 16x32 f16: lane&15 = M row; lane>>4 selects K {0..15} vs {16..31}
//  B 32x16 f16 (col-major src): lane&15 = N col; lane>>4 selects K {0..15} vs {16..31}
//  C 16x16 f32: VGPR r -> M = 8*(lane>>4) + r, N = lane&15
__global__ __launch_bounds__(256) void gemm_wmma_f16(
    const _Float16* __restrict__ A, const _Float16* __restrict__ Bt,
    const float* __restrict__ bias, float* __restrict__ C,
    int M, int K, int Ncols)
{
  const int lane = threadIdx.x & 31;
  const int wave = threadIdx.x >> 5;
  const int tilesN4 = Ncols >> 6;                 // groups of 4 N-tiles
  const int totalTiles = (M >> 4) * tilesN4;
  int tile = blockIdx.x * (blockDim.x >> 5) + wave;
  if (tile >= totalTiles) return;                 // wave-uniform guard: EXEC stays all-ones
  const int tm = tile / tilesN4;
  const int tg = tile - tm * tilesN4;
  const int row  = tm * 16 + (lane & 15);
  const int col0 = tg * 64 + (lane & 15);
  const int kbA = (lane >> 4) * 16;
  const int kbB = (lane >> 4) * 16;
  const _Float16* ap = A  + (size_t)row  * K + kbA;
  const _Float16* bp = Bt + (size_t)col0 * K + kbB;
  const size_t bstride = (size_t)16 * K;          // 16 columns ahead

  v8f acc0 = {}, acc1 = {}, acc2 = {}, acc3 = {};
  for (int k0 = 0; k0 < K; k0 += 32) {
    v16h a, b0, b1, b2, b3;
    ((float4*)&a)[0]  = *(const float4*)(ap + k0);
    ((float4*)&a)[1]  = *(const float4*)(ap + k0 + 8);
    const _Float16* bk = bp + k0;
    ((float4*)&b0)[0] = *(const float4*)(bk);
    ((float4*)&b0)[1] = *(const float4*)(bk + 8);
    ((float4*)&b1)[0] = *(const float4*)(bk + bstride);
    ((float4*)&b1)[1] = *(const float4*)(bk + bstride + 8);
    ((float4*)&b2)[0] = *(const float4*)(bk + 2 * bstride);
    ((float4*)&b2)[1] = *(const float4*)(bk + 2 * bstride + 8);
    ((float4*)&b3)[0] = *(const float4*)(bk + 3 * bstride);
    ((float4*)&b3)[1] = *(const float4*)(bk + 3 * bstride + 8);
    acc0 = __builtin_amdgcn_wmma_f32_16x16x32_f16(false, a, false, b0, (short)0, acc0, false, false);
    acc1 = __builtin_amdgcn_wmma_f32_16x16x32_f16(false, a, false, b1, (short)0, acc1, false, false);
    acc2 = __builtin_amdgcn_wmma_f32_16x16x32_f16(false, a, false, b2, (short)0, acc2, false, false);
    acc3 = __builtin_amdgcn_wmma_f32_16x16x32_f16(false, a, false, b3, (short)0, acc3, false, false);
  }
  const int rbase = tm * 16 + ((lane >> 4) << 3);
  v8f accs[4] = {acc0, acc1, acc2, acc3};
#pragma unroll
  for (int t = 0; t < 4; ++t) {
    const int col = col0 + 16 * t;
    const float badd = bias ? bias[col] : 0.0f;
#pragma unroll
    for (int r = 0; r < 8; ++r)
      C[(size_t)(rbase + r) * Ncols + col] = accs[t][r] + badd;
  }
}

// ---------- attention logits: Ll/Lr[n,h] = dot(z[n,:], a_{l,r}[h,:]) ----------
__global__ __launch_bounds__(256) void attn_logits(
    const float* __restrict__ z, const float* __restrict__ al, const float* __restrict__ ar,
    float* __restrict__ Ll, float* __restrict__ Lr)
{
  const int total = NN * HH;
  for (int i = blockIdx.x * blockDim.x + threadIdx.x; i < total; i += gridDim.x * blockDim.x) {
    int h = i & (HH - 1), n = i >> 1;
    const float4* zp  = (const float4*)(z  + (size_t)n * DD);
    const float4* alp = (const float4*)(al + (size_t)h * DD);
    const float4* arp = (const float4*)(ar + (size_t)h * DD);
    float sl = 0.f, sr = 0.f;
    for (int d = 0; d < DD / 4; ++d) {
      float4 x = zp[d], wl = alp[d], wr = arp[d];
      sl += x.x * wl.x + x.y * wl.y + x.z * wl.z + x.w * wl.w;
      sr += x.x * wr.x + x.y * wr.y + x.z * wr.z + x.w * wr.w;
    }
    Ll[i] = sl; Lr[i] = sr;
  }
}

// ---------- edge pass 1: e = lrelu(Ll[src]+Lr[dst]); segment max over dst ----------
__global__ __launch_bounds__(256) void edge_logit_max(
    const int* __restrict__ src, const int* __restrict__ dst,
    const float* __restrict__ Ll, const float* __restrict__ Lr,
    float* __restrict__ ebuf, unsigned* __restrict__ maxb)
{
  const int total = EE * HH;
  for (int i = blockIdx.x * blockDim.x + threadIdx.x; i < total; i += gridDim.x * blockDim.x) {
    int h = i & 1, e = i >> 1;
    int s = src[e], d = dst[e];
    float x = Ll[s * HH + h] + Lr[d * HH + h];
    x = (x > 0.f) ? x : NEG_ATT * x;
    ebuf[i] = x;
    atomicMax(&maxb[d * HH + h], f2o(x));
  }
}

// ---------- edge pass 2: ex = exp(e - max[dst]); segment sum ----------
__global__ __launch_bounds__(256) void edge_exp_sum(
    const int* __restrict__ dst,
    float* __restrict__ ebuf, const unsigned* __restrict__ maxb,
    float* __restrict__ denom)
{
  const int total = EE * HH;
  for (int i = blockIdx.x * blockDim.x + threadIdx.x; i < total; i += gridDim.x * blockDim.x) {
    int h = i & 1, e = i >> 1;
    int d = dst[e];
    float ex = expf(ebuf[i] - o2f(maxb[d * HH + h]));
    ebuf[i] = ex;
    atomicAdd(&denom[d * HH + h], ex);
  }
}

// ---------- edge pass 3: agg[dst,h,:] += alpha * wvn[src,h,:] (float4 gather, 4x atomicAdd) ----
__global__ __launch_bounds__(256) void edge_aggregate(
    const int* __restrict__ src, const int* __restrict__ dst,
    const float* __restrict__ ebuf, const float* __restrict__ denom,
    const float* __restrict__ wvn, float* __restrict__ agg)
{
  const int total = EE * HH * (DD / 4);           // 15.36M
  for (int i = blockIdx.x * blockDim.x + threadIdx.x; i < total; i += gridDim.x * blockDim.x) {
    int q  = i & (DD / 4 - 1);                    // 0..31
    int eh = i >> 5;
    int h = eh & 1, e = eh >> 1;
    int s = src[e], dn = dst[e];
    float alpha = ebuf[eh] / denom[dn * HH + h];
    const int off = h * DD + q * 4;
    float4 w = *(const float4*)(wvn + (size_t)s * HH * DD + off);
    float* ap = agg + (size_t)dn * HH * DD + off;
    atomicAdd(ap + 0, alpha * w.x);
    atomicAdd(ap + 1, alpha * w.y);
    atomicAdd(ap + 2, alpha * w.z);
    atomicAdd(ap + 3, alpha * w.w);
  }
}

// ---------- edge pass 4: mean-sum + degree (float4 gather) ----------
__global__ __launch_bounds__(256) void edge_mean_deg(
    const int* __restrict__ src, const int* __restrict__ dst,
    const float* __restrict__ v, float* __restrict__ meansum, float* __restrict__ deg)
{
  const int total = EE * (DD / 4);                // 7.68M
  for (int i = blockIdx.x * blockDim.x + threadIdx.x; i < total; i += gridDim.x * blockDim.x) {
    int q = i & (DD / 4 - 1), e = i >> 5;
    int s = src[e], dn = dst[e];
    float4 x = *(const float4*)(v + (size_t)s * DD + q * 4);
    float* mp = meansum + (size_t)dn * DD + q * 4;
    atomicAdd(mp + 0, x.x);
    atomicAdd(mp + 1, x.y);
    atomicAdd(mp + 2, x.z);
    atomicAdd(mp + 3, x.w);
    if (q == 0) atomicAdd(&deg[dn], 1.0f);
  }
}

// ---------- edge pass 5: segment max of (v@Wgm)[src] (float4 gather) ----------
__global__ __launch_bounds__(256) void edge_max_gm(
    const int* __restrict__ src, const int* __restrict__ dst,
    const float* __restrict__ vgm, unsigned* __restrict__ mxb)
{
  const int total = EE * (DMG / 4);               // 3.84M
  for (int i = blockIdx.x * blockDim.x + threadIdx.x; i < total; i += gridDim.x * blockDim.x) {
    int q = i & (DMG / 4 - 1), e = i >> 4;
    float4 x = *(const float4*)(vgm + (size_t)src[e] * DMG + q * 4);
    unsigned* mp = mxb + (size_t)dst[e] * DMG + q * 4;
    atomicMax(mp + 0, f2o(x.x));
    atomicMax(mp + 1, f2o(x.y));
    atomicMax(mp + 2, f2o(x.z));
    atomicMax(mp + 3, f2o(x.w));
  }
}

// ---------- per-node finalize: gate + merge + leaky_relu ----------
__global__ __launch_bounds__(256) void finalize_node(
    const float* __restrict__ v, const float* __restrict__ meansum,
    const float* __restrict__ deg, const unsigned* __restrict__ mxb,
    const float* __restrict__ agg, const float* __restrict__ Wg /*[320][H]*/,
    const float* __restrict__ bg /*[H]*/, float* __restrict__ outp, int merge_cat)
{
  for (int n = blockIdx.x * blockDim.x + threadIdx.x; n < NN; n += gridDim.x * blockDim.x) {
    float dgv = deg[n];
    float invd = 1.0f / fmaxf(dgv, 1.0f);
    float a0 = bg[0], a1 = bg[1];
    const float4* vp = (const float4*)(v + (size_t)n * DD);
    const float4* mp = (const float4*)(meansum + (size_t)n * DD);
    for (int d = 0; d < DD / 4; ++d) {
      float4 x = vp[d];
      const float* w = Wg + (d * 4) * HH;
      a0 += x.x * w[0] + x.y * w[2] + x.z * w[4] + x.w * w[6];
      a1 += x.x * w[1] + x.y * w[3] + x.z * w[5] + x.w * w[7];
    }
    for (int d = 0; d < DD / 4; ++d) {
      float4 x = mp[d];
      const float* w = Wg + (DD + d * 4) * HH;
      a0 += invd * (x.x * w[0] + x.y * w[2] + x.z * w[4] + x.w * w[6]);
      a1 += invd * (x.x * w[1] + x.y * w[3] + x.z * w[5] + x.w * w[7]);
    }
    const unsigned* xp = mxb + (size_t)n * DMG;
    for (int d = 0; d < DMG; ++d) {
      float x = (dgv > 0.0f) ? o2f(xp[d]) : 0.0f;
      a0 += x * Wg[(2 * DD + d) * HH + 0];
      a1 += x * Wg[(2 * DD + d) * HH + 1];
    }
    float g0 = 1.0f / (1.0f + expf(-a0));
    float g1 = 1.0f / (1.0f + expf(-a1));
    const float* ag = agg + (size_t)n * HH * DD;
    if (merge_cat) {
      float* op = outp + (size_t)n * HH * DD;
      for (int d = 0; d < DD; ++d) {
        float y0 = g0 * ag[d];      op[d]      = (y0 > 0.f) ? y0 : NEG_OUT * y0;
        float y1 = g1 * ag[DD + d]; op[DD + d] = (y1 > 0.f) ? y1 : NEG_OUT * y1;
      }
    } else {
      float* op = outp + (size_t)n * DD;
      for (int d = 0; d < DD; ++d) {
        float y = 0.5f * (g0 * ag[d] + g1 * ag[DD + d]);
        op[d] = (y > 0.f) ? y : NEG_OUT * y;
      }
    }
  }
}

// ---------- concat [proj | cur0 | cur1] -> f16 [N, 640] ----------
__global__ __launch_bounds__(256) void concat_h16(
    const float* __restrict__ proj, const float* __restrict__ c0,
    const float* __restrict__ c1, _Float16* __restrict__ out)
{
  const int total = NN * CONCATK;
  for (int i = blockIdx.x * blockDim.x + threadIdx.x; i < total; i += gridDim.x * blockDim.x) {
    int c = i % CONCATK, n = i / CONCATK;
    float x;
    if (c < DD)              x = proj[(size_t)n * DD + c];
    else if (c < DD + HH*DD) x = c0[(size_t)n * HH * DD + (c - DD)];
    else                     x = c1[(size_t)n * HH * DD + (c - DD - HH * DD)];
    out[i] = (_Float16)x;
  }
}

// ---------- host orchestration ----------
static inline int gsblocks(long n) {
  long b = (n + 255) / 256;
  if (b > 16384) b = 16384;
  if (b < 1) b = 1;
  return (int)b;
}

extern "C" void kernel_launch(void* const* d_in, const int* in_sizes, int n_in,
                              void* d_out, int out_size, void* d_ws, size_t ws_size,
                              hipStream_t stream) {
  (void)in_sizes; (void)n_in; (void)out_size; (void)ws_size;
  const float* ft      = (const float*)d_in[0];
  const int*   src     = (const int*)d_in[1];   // [V,E]
  const int*   dst     = (const int*)d_in[2];   // [V,E]
  const float* W_proj  = (const float*)d_in[3]; // [IN,D]
  const float* W_embed = (const float*)d_in[4]; // [D,D]
  const float* a_l     = (const float*)d_in[5]; // [V,BLK,H,D]
  const float* a_r     = (const float*)d_in[6];
  const float* Wv      = (const float*)d_in[7]; // [V,BLK,H,D,D]
  const float* Wgm     = (const float*)d_in[8]; // [V,BLK,D,DM]
  const float* Wgate   = (const float*)d_in[9]; // [V,BLK,320,H]
  const float* b_gate  = (const float*)d_in[10];// [V,BLK,H]
  const float* W_tran  = (const float*)d_in[11];// [640,OUT]
  const float* b_tran  = (const float*)d_in[12];// [OUT]
  float* out = (float*)d_out;

  // bump arena over d_ws (256B aligned chunks)
  char* p = (char*)d_ws;
  auto alloc = [&](size_t bytes) -> char* {
    char* r = p; p += (bytes + 255) & ~(size_t)255; return r;
  };

  _Float16* ft16   = (_Float16*)alloc((size_t)NN * IND * 2);
  _Float16* Wp_t   = (_Float16*)alloc((size_t)DD * IND * 2);                 // [128][512]
  _Float16* We_t   = (_Float16*)alloc((size_t)DD * DD * 2);                  // [128][128]
  _Float16* Wv_t   = (_Float16*)alloc((size_t)VV * BLKN * HH * DD * DD * 2); // per (j,i): [256][128]
  _Float16* Wgm_t  = (_Float16*)alloc((size_t)VV * BLKN * DMG * DD * 2);     // per (j,i): [64][128]
  _Float16* Wt_t   = (_Float16*)alloc((size_t)OUTD * CONCATK * 2);           // [64][640]
  float* proj      = (float*)alloc((size_t)NN * DD * 4);
  float* l0out0    = (float*)alloc((size_t)NN * DD * 4);
  float* l0out1    = (float*)alloc((size_t)NN * DD * 4);
  float* l1out0    = (float*)alloc((size_t)NN * HH * DD * 4);
  float* l1out1    = (float*)alloc((size_t)NN * HH * DD * 4);
  _Float16* v16    = (_Float16*)alloc((size_t)NN * DD * 2);
  float* z         = (float*)alloc((size_t)NN * DD * 4);  // reused as meansum after logits
  float* Ll        = (float*)alloc((size_t)NN * HH * 4);
  float* Lr        = (float*)alloc((size_t)NN * HH * 4);
  unsigned* maxb   = (unsigned*)alloc((size_t)NN * HH * 4);
  float* denom     = (float*)alloc((size_t)NN * HH * 4);
  float* ebuf      = (float*)alloc((size_t)EE * HH * 4);
  float* wvn       = (float*)alloc((size_t)NN * HH * DD * 4);
  float* agg       = (float*)alloc((size_t)NN * HH * DD * 4);
  float* vgm       = (float*)alloc((size_t)NN * DMG * 4);
  unsigned* mxb    = (unsigned*)alloc((size_t)NN * DMG * 4);
  float* deg       = (float*)alloc((size_t)NN * 4);
  // concat buffer aliases wvn+agg (both dead after last finalize)
  _Float16* h16    = (_Float16*)wvn;   // needs NN*640*2 = 38.4MB < wvn+agg = 61.4MB
  float* meansum   = z;                // z dead after attn_logits

  // 16x64 strip per wave, 8 waves per block
  auto gemm_blocks = [](int M, int Nc) { return ((M / 16) * (Nc / 64) + 7) / 8; };

  // ---- weight prep (cheap, deterministic every call) ----
  cvt_f32_f16<<<gsblocks((long)NN * IND), 256, 0, stream>>>(ft, ft16, NN * IND);
  tconv<<<gsblocks((long)IND * DD), 256, 0, stream>>>(W_proj, Wp_t, IND, DD);
  tconv<<<gsblocks((long)DD * DD), 256, 0, stream>>>(W_embed, We_t, DD, DD);
  for (int b = 0; b < VV * BLKN * HH; ++b)   // per-head DxD transposes -> [h*D+e][d]
    tconv<<<gsblocks((long)DD * DD), 256, 0, stream>>>(
        Wv + (size_t)b * DD * DD, Wv_t + (size_t)b * DD * DD, DD, DD);
  for (int b = 0; b < VV * BLKN; ++b)
    tconv<<<gsblocks((long)DD * DMG), 256, 0, stream>>>(
        Wgm + (size_t)b * DD * DMG, Wgm_t + (size_t)b * DMG * DD, DD, DMG);
  tconv<<<gsblocks((long)CONCATK * OUTD), 256, 0, stream>>>(W_tran, Wt_t, CONCATK, OUTD);

  // ---- proj = ft @ W_proj ----
  gemm_wmma_f16<<<gemm_blocks(NN, DD), 256, 0, stream>>>(ft16, Wp_t, nullptr, proj, NN, IND, DD);

  float* l0out[2] = {l0out0, l0out1};
  float* l1out[2] = {l1out0, l1out1};

  for (int layer = 0; layer < BLKN; ++layer) {
    for (int j = 0; j < VV; ++j) {
      const float* vin = (layer == 0) ? proj : l0out[j];
      float* vout = (layer == 0) ? l0out[j] : l1out[j];
      const int ji = j * BLKN + layer;
      const int* sj = src + (size_t)j * EE;
      const int* dj = dst + (size_t)j * EE;

      // f16 copy of layer input (A-operand for all GEMMs this layer-view)
      cvt_f32_f16<<<gsblocks((long)NN * DD), 256, 0, stream>>>(vin, v16, NN * DD);

      // z = v @ W_embed ; attention logits
      gemm_wmma_f16<<<gemm_blocks(NN, DD), 256, 0, stream>>>(v16, We_t, nullptr, z, NN, DD, DD);
      attn_logits<<<gsblocks((long)NN * HH), 256, 0, stream>>>(
          z, a_l + (size_t)ji * HH * DD, a_r + (size_t)ji * HH * DD, Ll, Lr);

      // init accumulators (AFTER logits: meansum aliases z)
      fill_u32<<<gsblocks((long)NN * HH), 256, 0, stream>>>(maxb, ORD_NEG_INF, NN * HH);
      fill_u32<<<gsblocks((long)NN * HH), 256, 0, stream>>>((unsigned*)denom, 0u, NN * HH);
      fill_u32<<<gsblocks((long)NN * HH * DD), 256, 0, stream>>>((unsigned*)agg, 0u, NN * HH * DD);
      fill_u32<<<gsblocks((long)NN * DD), 256, 0, stream>>>((unsigned*)meansum, 0u, NN * DD);
      fill_u32<<<gsblocks((long)NN), 256, 0, stream>>>((unsigned*)deg, 0u, NN);
      fill_u32<<<gsblocks((long)NN * DMG), 256, 0, stream>>>(mxb, ORD_NEG_INF, NN * DMG);

      // edge softmax
      edge_logit_max<<<gsblocks((long)EE * HH), 256, 0, stream>>>(sj, dj, Ll, Lr, ebuf, maxb);
      edge_exp_sum<<<gsblocks((long)EE * HH), 256, 0, stream>>>(dj, ebuf, maxb, denom);

      // per-head value projection (both heads as one 256-col GEMM) and gate max-pool proj
      gemm_wmma_f16<<<gemm_blocks(NN, HH * DD), 256, 0, stream>>>(
          v16, Wv_t + (size_t)ji * HH * DD * DD, nullptr, wvn, NN, DD, HH * DD);
      gemm_wmma_f16<<<gemm_blocks(NN, DMG), 256, 0, stream>>>(
          v16, Wgm_t + (size_t)ji * DMG * DD, nullptr, vgm, NN, DD, DMG);

      // scatter aggregations (b128 gathers + dword atomics)
      edge_aggregate<<<gsblocks((long)EE * HH * (DD / 4)), 256, 0, stream>>>(
          sj, dj, ebuf, denom, wvn, agg);
      edge_mean_deg<<<gsblocks((long)EE * (DD / 4)), 256, 0, stream>>>(
          sj, dj, vin, meansum, deg);
      edge_max_gm<<<gsblocks((long)EE * (DMG / 4)), 256, 0, stream>>>(sj, dj, vgm, mxb);

      // gate + merge + leaky_relu
      finalize_node<<<gsblocks((long)NN), 256, 0, stream>>>(
          vin, meansum, deg, mxb, agg,
          Wgate + (size_t)ji * (2 * DD + DMG) * HH, b_gate + (size_t)ji * HH,
          vout, (layer == BLKN - 1) ? 1 : 0);
    }
  }

  // final: out = [proj | cur0 | cur1] @ W_tran + b_tran
  concat_h16<<<gsblocks((long)NN * CONCATK), 256, 0, stream>>>(proj, l1out[0], l1out[1], h16);
  gemm_wmma_f16<<<gemm_blocks(NN, OUTD), 256, 0, stream>>>(h16, Wt_t, b_tran, out, NN, CONCATK, OUTD);
}
